// QuantizedModel_22162031247829
// MI455X (gfx1250) — matrix operand order repaired and checked by reading, not compiled, not run on hardware
//
#include <hip/hip_runtime.h>

typedef __attribute__((ext_vector_type(16))) _Float16 v16h;
typedef __attribute__((ext_vector_type(8)))  _Float16 v8h;
typedef __attribute__((ext_vector_type(4)))  _Float16 v4h;
typedef __attribute__((ext_vector_type(8)))  float    v8f;
typedef __attribute__((ext_vector_type(4)))  float    v4f;
typedef __attribute__((ext_vector_type(4)))  int      v4i;

#define MT 128      // M tile
#define NT 128      // N tile
#define KT 64       // K tile (GROUP=128 => scale/zero constant per K tile)
#define LDST 72     // LDS row stride in halves: 144B = 9*16B, 16B-aligned, bank-rotating

__device__ __forceinline__ v8f wmma16(v16h a, v16h b, v8f c) {
  // v_wmma_f32_16x16x32_f16: (neg_a, A, neg_b, B, c_mod, C, reuse_a, reuse_b)
  return __builtin_amdgcn_wmma_f32_16x16x32_f16(false, a, false, b, (short)0, c,
                                                false, false);
}

// A fragment (16x32 f16): lane m = lane&15; VGPR0..7 hold K=kb..kb+15
// with kb = (lane>>4)*16  -> two 16B LDS loads.
__device__ __forceinline__ v16h ldsA(const _Float16* t, int rbase, int kk, int lane) {
  int r  = rbase + (lane & 15);
  int kb = kk + ((lane >> 4) << 4);
  const _Float16* p = t + r * LDST + kb;
  union { v16h v; v8h h[2]; } u;
  u.h[0] = *(const v8h*)(p);
  u.h[1] = *(const v8h*)(p + 8);
  return u.v;
}

// B fragment (32x16 f16): lane n = lane&15; lanes 0-15 hold K=kk..kk+15,
// lanes 16-31 hold K=kk+16..kk+31 (contiguous 16 halves per lane).
__device__ __forceinline__ v16h ldsB(const _Float16* t, int rbase, int kk, int lane) {
  int r  = rbase + (lane & 15);
  int kb = kk + ((lane >> 4) << 4);
  const _Float16* p = t + r * LDST + kb;
  union { v16h v; v8h h[2]; } u;
  u.h[0] = *(const v8h*)(p);
  u.h[1] = *(const v8h*)(p + 8);
  return u.v;
}

// C[m,n] = act( sum_k A[m,k] * dequant(Q[n,k]) + bias[n] )
// A row-major [M,K] (f32 or f16), Q row-major [N,K] int32 in [0,15],
// S/Z per [N, K/128]. One workgroup computes a 128x128 C tile.
// Software-pipelined: global loads for step k+1 issue before the WMMAs of
// step k so HBM latency overlaps matrix work.
template<bool A_F16, bool RELU, bool OUT_F16>
__global__ __launch_bounds__(256) void mlp_int4_wmma(
    const void* __restrict__ Aptr, const int* __restrict__ Q,
    const float* __restrict__ S, const float* __restrict__ Zp,
    const float* __restrict__ bias, void* __restrict__ Cptr,
    int M, int N, int K)
{
  __shared__ _Float16 As[MT * LDST];
  __shared__ _Float16 Bs[NT * LDST];

  const int tid  = threadIdx.x;
  const int lane = tid & 31;
  const int wave = tid >> 5;
  const int wm   = wave & 3;        // 4 waves along M (4 * 32 = 128)
  const int wn   = wave >> 2;       // 2 waves along N (2 * 64 = 128)
  const int m0   = blockIdx.y * MT;
  const int n0   = blockIdx.x * NT;
  const int KG   = K >> 7;          // quant groups per row (GROUP = 128)

  v8f acc[2][4] = {};

  // register staging for the next K-step (double buffer vs LDS)
  v4f   aF[8];                      // f32 activation path (layer 1)
  v8h   aH[4];                      // f16 activation path (layer 2)
  v4i   qv[8];
  float sv[8], zv[8];

  auto load_stage = [&](int k0) {
    if constexpr (A_F16) {
      const _Float16* A = (const _Float16*)Aptr;
      #pragma unroll
      for (int it = 0; it < 4; ++it) {
        int idx = tid + it * 256;
        int r = idx >> 3, c = (idx & 7) << 3;
        aH[it] = *(const v8h*)(A + (size_t)(m0 + r) * K + k0 + c);
      }
    } else {
      const float* A = (const float*)Aptr;
      #pragma unroll
      for (int it = 0; it < 8; ++it) {
        int idx = tid + it * 256;
        int r = idx >> 4, c = (idx & 15) << 2;
        aF[it] = *(const v4f*)(A + (size_t)(m0 + r) * K + k0 + c);
      }
    }
    const int kg = k0 >> 7;
    #pragma unroll
    for (int it = 0; it < 8; ++it) {
      int idx = tid + it * 256;
      int r = idx >> 4, c = (idx & 15) << 2;
      int n = n0 + r;
      qv[it] = *(const v4i*)(Q + (size_t)n * K + k0 + c);
      sv[it] = S [(size_t)n * KG + kg];
      zv[it] = Zp[(size_t)n * KG + kg];
    }
  };

  auto commit_stage = [&]() {
    if constexpr (A_F16) {
      #pragma unroll
      for (int it = 0; it < 4; ++it) {
        int idx = tid + it * 256;
        int r = idx >> 3, c = (idx & 7) << 3;
        *(v8h*)(&As[r * LDST + c]) = aH[it];
      }
    } else {
      #pragma unroll
      for (int it = 0; it < 8; ++it) {
        int idx = tid + it * 256;
        int r = idx >> 4, c = (idx & 15) << 2;
        v4h h;
        h.x = (_Float16)aF[it].x; h.y = (_Float16)aF[it].y;
        h.z = (_Float16)aF[it].z; h.w = (_Float16)aF[it].w;
        *(v4h*)(&As[r * LDST + c]) = h;
      }
    }
    #pragma unroll
    for (int it = 0; it < 8; ++it) {
      int idx = tid + it * 256;
      int r = idx >> 4, c = (idx & 15) << 2;
      float s = sv[it], z = zv[it];
      v4h h;
      h.x = (_Float16)(((float)qv[it].x - z) * s);
      h.y = (_Float16)(((float)qv[it].y - z) * s);
      h.z = (_Float16)(((float)qv[it].z - z) * s);
      h.w = (_Float16)(((float)qv[it].w - z) * s);
      *(v4h*)(&Bs[r * LDST + c]) = h;
    }
  };

  load_stage(0);
  for (int k0 = 0; k0 < K; k0 += KT) {
    commit_stage();                 // regs (step k) -> LDS, convert/dequant
    __syncthreads();
    if (k0 + KT < K) load_stage(k0 + KT);   // prefetch step k+1 into regs

    #pragma unroll
    for (int kk = 0; kk < KT; kk += 32) {
      v16h a0 = ldsA(As, wm * 32,      kk, lane);
      v16h a1 = ldsA(As, wm * 32 + 16, kk, lane);
      v16h b0 = ldsB(Bs, wn * 64,      kk, lane);
      v16h b1 = ldsB(Bs, wn * 64 + 16, kk, lane);
      v16h b2 = ldsB(Bs, wn * 64 + 32, kk, lane);
      v16h b3 = ldsB(Bs, wn * 64 + 48, kk, lane);
      acc[0][0] = wmma16(a0, b0, acc[0][0]);
      acc[0][1] = wmma16(a0, b1, acc[0][1]);
      acc[0][2] = wmma16(a0, b2, acc[0][2]);
      acc[0][3] = wmma16(a0, b3, acc[0][3]);
      acc[1][0] = wmma16(a1, b0, acc[1][0]);
      acc[1][1] = wmma16(a1, b1, acc[1][1]);
      acc[1][2] = wmma16(a1, b2, acc[1][2]);
      acc[1][3] = wmma16(a1, b3, acc[1][3]);
    }
    __syncthreads();
  }

  // ---- epilogue: bias (+ReLU); C-frag layout: n = lane&15, m = r + 8*(lane>>4) ----
  const int cn   = lane & 15;
  const int rofs = (lane >> 4) << 3;
  #pragma unroll
  for (int j = 0; j < 4; ++j) {
    int n = n0 + wn * 64 + j * 16 + cn;
    float bv = bias[n];
    #pragma unroll
    for (int i = 0; i < 2; ++i) {
      int mbase = m0 + wm * 32 + i * 16 + rofs;
      #pragma unroll
      for (int r = 0; r < 8; ++r) {
        float v = acc[i][j][r] + bv;
        if (RELU) v = fmaxf(v, 0.0f);
        size_t off = (size_t)(mbase + r) * N + n;
        if (OUT_F16) ((_Float16*)Cptr)[off] = (_Float16)v;
        else         ((float*)Cptr)[off]    = v;
      }
    }
  }
}

extern "C" void kernel_launch(void* const* d_in, const int* in_sizes, int n_in,
                              void* d_out, int out_size, void* d_ws, size_t ws_size,
                              hipStream_t stream) {
  const float* x  = (const float*)d_in[0];
  const int*   q1 = (const int*)  d_in[1];
  const float* s1 = (const float*)d_in[2];
  const float* z1 = (const float*)d_in[3];
  const float* b1 = (const float*)d_in[4];
  const int*   q2 = (const int*)  d_in[5];
  const float* s2 = (const float*)d_in[6];
  const float* z2 = (const float*)d_in[7];
  const float* b2 = (const float*)d_in[8];

  const int DH   = in_sizes[4];            // 8192
  const int DOUT = in_sizes[8];            // 2048
  const int DIN  = in_sizes[1] / DH;       // 2048
  const int B    = in_sizes[0] / DIN;      // 512

  _Float16* h = (_Float16*)d_ws;           // B x DH f16 intermediate (8 MB)

  dim3 blk(256);
  dim3 g1(DH / NT, B / MT);
  mlp_int4_wmma<false, true,  true ><<<g1, blk, 0, stream>>>(
      x, q1, s1, z1, b1, (void*)h, B, DH, DIN);

  dim3 g2(DOUT / NT, B / MT);
  mlp_int4_wmma<true,  false, false><<<g2, blk, 0, stream>>>(
      (const void*)h, q2, s2, z2, b2, d_out, B, DOUT, DH);
}
